// MoE_3006477107310
// MI455X (gfx1250) — compile-verified, matrix operand order checked by reading, not run on hardware
//
#include <hip/hip_runtime.h>
#include <hip/hip_bf16.h>
#include <math.h>

// ---------- types for WMMA ----------
typedef __bf16 bf16_t;
typedef bf16_t v16bf __attribute__((ext_vector_type(16)));
typedef bf16_t v8bf  __attribute__((ext_vector_type(8)));
typedef bf16_t v2bf  __attribute__((ext_vector_type(2)));
typedef float  v8f   __attribute__((ext_vector_type(8)));

// ---------- problem constants ----------
#define T_TOK 2048
#define HDIM  1024
#define IDIM  512
#define EDIM  8
#define SHI   1024
#define KCOLS (EDIM*IDIM + SHI)   // 5120

// ---------- GEMM tiling ----------
#define BM 64
#define BN 128
#define KT 32            // one wmma_f32_16x16x32_bf16 K-step
#define LPITCH 40        // 80B LDS rows -> 16B-aligned b128 frags, conflict-free stride

__device__ __forceinline__ float sigmoid_(float x) {
    return __builtin_amdgcn_rcpf(1.0f + __expf(-x));
}
__device__ __forceinline__ float silu_(float g) {
    return g * __builtin_amdgcn_rcpf(1.0f + __expf(-g));
}

// ============================================================
// Kernel 0: gating. One wave per token -> mask_w[T,E]
// ============================================================
__global__ void __launch_bounds__(256)
gate_kernel(const float* __restrict__ x,
            const float* __restrict__ gate_w,    // [H, E]
            const float* __restrict__ gate_bias, // [E]
            float* __restrict__ mask_w)          // [T, E]
{
    int wave = threadIdx.x >> 5;
    int lane = threadIdx.x & 31;
    int t = blockIdx.x * 8 + wave;
    if (t >= T_TOK) return;

    float acc[EDIM];
#pragma unroll
    for (int e = 0; e < EDIM; ++e) acc[e] = 0.0f;

    const float* xr = x + (size_t)t * HDIM;
    for (int h = lane; h < HDIM; h += 32) {
        float xv = xr[h];
        const float* gw = gate_w + (size_t)h * EDIM;
#pragma unroll
        for (int e = 0; e < EDIM; ++e) acc[e] += xv * gw[e];
    }
#pragma unroll
    for (int e = 0; e < EDIM; ++e) {
#pragma unroll
        for (int off = 16; off > 0; off >>= 1)
            acc[e] += __shfl_xor(acc[e], off, 32);
    }

    if (lane == 0) {
        float sfc[EDIM];
#pragma unroll
        for (int e = 0; e < EDIM; ++e) sfc[e] = sigmoid_(acc[e]) + gate_bias[e];

        float gs[4];
#pragma unroll
        for (int g = 0; g < 4; ++g) gs[g] = sfc[2 * g] + sfc[2 * g + 1];

        int g1 = 0;
        for (int g = 1; g < 4; ++g) if (gs[g] > gs[g1]) g1 = g;
        int g2 = -1;
        for (int g = 0; g < 4; ++g) { if (g == g1) continue; if (g2 < 0 || gs[g] > gs[g2]) g2 = g; }

        float b1 = -1e30f, b2 = -1e30f; int e1 = 0, e2 = 0;
        for (int g = 0; g < 4; ++g) {
            if (g != g1 && g != g2) continue;
            for (int j = 0; j < 2; ++j) {
                int e = 2 * g + j; float v = sfc[e];
                if (v > b1)      { b2 = b1; e2 = e1; b1 = v; e1 = e; }
                else if (v > b2) { b2 = v;  e2 = e; }
            }
        }
        float denom = b1 + b2 + 1e-20f;
        float* mrow = mask_w + (size_t)t * EDIM;
#pragma unroll
        for (int e = 0; e < EDIM; ++e) mrow[e] = 0.0f;
        mrow[e1] = b1 / denom * 2.5f;
        mrow[e2] = b2 / denom * 2.5f;
    }
}

// ============================================================
// WMMA fragment load from LDS (ISA 7.12.2 16-bit A layout).
// ============================================================
union FragU { v8bf h[2]; v16bf v; };

__device__ __forceinline__ v16bf load_frag(const bf16_t* base)
{
    int lane = threadIdx.x & 31;
    int row  = lane & 15;
    int kh   = (lane >> 4) * 8;
    FragU u;
    u.h[0] = *(const v8bf*)(base + row * LPITCH + kh);
    u.h[1] = *(const v8bf*)(base + row * LPITCH + kh + 16);
    return u.v;
}

// ============================================================
// Kernel 1: fused gate+up. Templated on routed/shared so the weight
// leading dimension is a compile-time constant; staging addresses are
// (scalar base advanced per K-step) + (thread-constant 32-bit offset).
// ============================================================
template<bool IS_SHARED>
__global__ void __launch_bounds__(256)
moe_gateup_kernel(const float* __restrict__ x,      // [T, H]
                  const float* __restrict__ Wg,     // routed: [E,H,I]; shared: [H,SHI]
                  const float* __restrict__ Wu,
                  const float* __restrict__ mask_w, // [T, E]
                  bf16_t* __restrict__ Abuf)        // [T, KCOLS]
{
    constexpr int LDB = IS_SHARED ? SHI : IDIM;

    __shared__ bf16_t lA [2][BM * LPITCH];
    __shared__ bf16_t lBg[2][BN * LPITCH];
    __shared__ bf16_t lBu[2][BN * LPITCH];

    int bm   = blockIdx.x & 31;
    int rest = blockIdx.x >> 5;

    const float *Bg, *Bu;
    int colBase, e = 0;
    if constexpr (IS_SHARED) {
        Bg = Wg + rest * BN;
        Bu = Wu + rest * BN;
        colBase = EDIM * IDIM + rest * BN;
    } else {
        e = rest >> 2;
        int bn = rest & 3;
        Bg = Wg + (size_t)e * HDIM * IDIM + bn * BN;
        Bu = Wu + (size_t)e * HDIM * IDIM + bn * BN;
        colBase = e * IDIM + bn * BN;
    }
    int m0 = bm * BM;

    int tid   = threadIdx.x;
    int wave  = tid >> 5;
    int waveM = wave & 1;
    int waveN = wave >> 1;

    // thread-constant offsets (elements)
    int aGlb[4], aLds[4];
#pragma unroll
    for (int i = 0; i < 4; ++i) {
        int p = tid + i * 256;
        int mm = p >> 4, kp = (p & 15) * 2;
        aGlb[i] = mm * HDIM + kp;
        aLds[i] = mm * LPITCH + kp;
    }
    int bGlb[8], bLds[8];
#pragma unroll
    for (int i = 0; i < 8; ++i) {
        int p = tid + i * 256;
        int nn = p & 127, kp2 = p >> 7;
        bGlb[i] = 2 * kp2 * LDB + nn;
        bLds[i] = nn * LPITCH + 2 * kp2;
    }

    auto stage = [&](int buf, int k0) {
        const float* bA = x + (size_t)m0 * HDIM + k0;
#pragma unroll
        for (int i = 0; i < 4; ++i) {
            float2 v = *(const float2*)(bA + aGlb[i]);
            v2bf pk; pk[0] = (bf16_t)v.x; pk[1] = (bf16_t)v.y;
            *(v2bf*)&lA[buf][aLds[i]] = pk;
        }
        const float* bG = Bg + (size_t)k0 * LDB;
        const float* bU = Bu + (size_t)k0 * LDB;
#pragma unroll
        for (int i = 0; i < 8; ++i) {
            v2bf g; g[0] = (bf16_t)bG[bGlb[i]]; g[1] = (bf16_t)bG[bGlb[i] + LDB];
            v2bf u; u[0] = (bf16_t)bU[bGlb[i]]; u[1] = (bf16_t)bU[bGlb[i] + LDB];
            *(v2bf*)&lBg[buf][bLds[i]] = g;
            *(v2bf*)&lBu[buf][bLds[i]] = u;
        }
        // prefetch weight stream 2 K-tiles ahead into GL2 (global_prefetch_b8)
        if (k0 + 2 * KT < HDIM) {
#pragma unroll
            for (int i = 0; i < 8; ++i) {
                const float* pf = ((i & 1) ? bU : bG) + bGlb[i] + (size_t)(2 * KT) * LDB;
                __builtin_prefetch(pf, 0, 1);
            }
        }
    };

    v8f cg[2][2] = {};
    v8f cu[2][2] = {};

    stage(0, 0);
    __syncthreads();

    const int nK = HDIM / KT;   // 32
    for (int kt = 0; kt < nK; ++kt) {
        int cur = kt & 1;
        if (kt + 1 < nK) stage(cur ^ 1, (kt + 1) * KT);

        v16bf a0 = load_frag(&lA[cur][(waveM * 32 + 0 ) * LPITCH]);
        v16bf a1 = load_frag(&lA[cur][(waveM * 32 + 16) * LPITCH]);
#pragma unroll
        for (int ni = 0; ni < 2; ++ni) {
            v16bf bg = load_frag(&lBg[cur][(waveN * 32 + ni * 16) * LPITCH]);
            v16bf bu = load_frag(&lBu[cur][(waveN * 32 + ni * 16) * LPITCH]);
            cg[0][ni] = __builtin_amdgcn_wmma_f32_16x16x32_bf16(false, a0, false, bg, (short)0, cg[0][ni], false, false);
            cg[1][ni] = __builtin_amdgcn_wmma_f32_16x16x32_bf16(false, a1, false, bg, (short)0, cg[1][ni], false, false);
            cu[0][ni] = __builtin_amdgcn_wmma_f32_16x16x32_bf16(false, a0, false, bu, (short)0, cu[0][ni], false, false);
            cu[1][ni] = __builtin_amdgcn_wmma_f32_16x16x32_bf16(false, a1, false, bu, (short)0, cu[1][ni], false, false);
        }
        __syncthreads();
    }

    // ---- epilogue ----
    int lane = tid & 31;
    int nl = lane & 15;
    int mh = (lane >> 4) * 8;

    float wfac[2][8];
    if constexpr (IS_SHARED) {
#pragma unroll
        for (int mi = 0; mi < 2; ++mi)
#pragma unroll
            for (int r = 0; r < 8; ++r) wfac[mi][r] = 1.0f;
    } else {
#pragma unroll
        for (int mi = 0; mi < 2; ++mi)
#pragma unroll
            for (int r = 0; r < 8; ++r) {
                int m = m0 + waveM * 32 + mi * 16 + mh + r;
                wfac[mi][r] = mask_w[(size_t)m * EDIM + e];
            }
    }

#pragma unroll
    for (int mi = 0; mi < 2; ++mi)
#pragma unroll
        for (int ni = 0; ni < 2; ++ni)
#pragma unroll
            for (int r = 0; r < 8; ++r) {
                int m = m0 + waveM * 32 + mi * 16 + mh + r;
                int n = colBase + waveN * 32 + ni * 16 + nl;
                float g = cg[mi][ni][r];
                float u = cu[mi][ni][r];
                Abuf[(size_t)m * KCOLS + n] = (bf16_t)(silu_(g) * u * wfac[mi][r]);
            }
}

// ============================================================
// Kernel 2: out[T,H] = Abuf[T,5120] @ concat(w_down [4096,H], sw_down [1024,H])
// ============================================================
__global__ void __launch_bounds__(256)
moe_down_kernel(const bf16_t* __restrict__ Abuf,
                const float* __restrict__ w_down,   // [E*I, H]
                const float* __restrict__ sw_down,  // [SHI, H]
                float* __restrict__ out)            // [T, H]
{
    __shared__ bf16_t lA[2][BM * LPITCH];
    __shared__ bf16_t lB[2][BN * LPITCH];

    int bm = blockIdx.x & 31;
    int bn = blockIdx.x >> 5;
    int m0 = bm * BM, n0 = bn * BN;

    int tid   = threadIdx.x;
    int wave  = tid >> 5;
    int waveM = wave & 1;
    int waveN = wave >> 1;

    // thread-constant offsets
    int aGlb = (tid >> 2) * KCOLS + (tid & 3) * 8;   // bf16 elements
    int aLds = (tid >> 2) * LPITCH + (tid & 3) * 8;
    int bGlb[8], bLds[8];
#pragma unroll
    for (int i = 0; i < 8; ++i) {
        int p = tid + i * 256;
        int nn = p & 127, kp2 = p >> 7;
        bGlb[i] = 2 * kp2 * HDIM + nn;
        bLds[i] = nn * LPITCH + 2 * kp2;
    }

    auto stage = [&](int buf, int k0) {
        const bf16_t* bA = Abuf + (size_t)m0 * KCOLS + k0;
        *(v8bf*)&lA[buf][aLds] = *(const v8bf*)(bA + aGlb);

        const float* base = (k0 < EDIM * IDIM)
                              ? (w_down  + (size_t)k0 * HDIM)
                              : (sw_down + (size_t)(k0 - EDIM * IDIM) * HDIM);
        base += n0;
#pragma unroll
        for (int i = 0; i < 8; ++i) {
            v2bf b; b[0] = (bf16_t)base[bGlb[i]]; b[1] = (bf16_t)base[bGlb[i] + HDIM];
            *(v2bf*)&lB[buf][bLds[i]] = b;
        }
        // prefetch 2 K-tiles ahead, staying within the same weight region
        int kpf = k0 + 2 * KT;
        if (kpf < KCOLS && ((k0 < EDIM * IDIM) == (kpf < EDIM * IDIM))) {
#pragma unroll
            for (int i = 0; i < 8; i += 2)
                __builtin_prefetch(base + bGlb[i] + (size_t)(2 * KT) * HDIM, 0, 1);
        }
    };

    v8f c[2][2] = {};

    stage(0, 0);
    __syncthreads();

    const int nK = KCOLS / KT;  // 160
    for (int kt = 0; kt < nK; ++kt) {
        int cur = kt & 1;
        if (kt + 1 < nK) stage(cur ^ 1, (kt + 1) * KT);

        v16bf a0 = load_frag(&lA[cur][(waveM * 32 + 0 ) * LPITCH]);
        v16bf a1 = load_frag(&lA[cur][(waveM * 32 + 16) * LPITCH]);
#pragma unroll
        for (int ni = 0; ni < 2; ++ni) {
            v16bf b = load_frag(&lB[cur][(waveN * 32 + ni * 16) * LPITCH]);
            c[0][ni] = __builtin_amdgcn_wmma_f32_16x16x32_bf16(false, a0, false, b, (short)0, c[0][ni], false, false);
            c[1][ni] = __builtin_amdgcn_wmma_f32_16x16x32_bf16(false, a1, false, b, (short)0, c[1][ni], false, false);
        }
        __syncthreads();
    }

    int lane = tid & 31;
    int nl = lane & 15;
    int mh = (lane >> 4) * 8;
#pragma unroll
    for (int mi = 0; mi < 2; ++mi)
#pragma unroll
        for (int ni = 0; ni < 2; ++ni)
#pragma unroll
            for (int r = 0; r < 8; ++r) {
                int m = m0 + waveM * 32 + mi * 16 + mh + r;
                int n = n0 + waveN * 32 + ni * 16 + nl;
                out[(size_t)m * HDIM + n] = c[mi][ni][r];
            }
}

// ============================================================
extern "C" void kernel_launch(void* const* d_in, const int* in_sizes, int n_in,
                              void* d_out, int out_size, void* d_ws, size_t ws_size,
                              hipStream_t stream)
{
    (void)in_sizes; (void)n_in; (void)out_size; (void)ws_size;

    const float* x         = (const float*)d_in[0];
    const float* gate_w    = (const float*)d_in[1];
    const float* gate_bias = (const float*)d_in[2];
    const float* w_gate    = (const float*)d_in[3];
    const float* w_up      = (const float*)d_in[4];
    const float* w_down    = (const float*)d_in[5];
    const float* sw_gate   = (const float*)d_in[6];
    const float* sw_up     = (const float*)d_in[7];
    const float* sw_down   = (const float*)d_in[8];

    float*  mask_w = (float*)d_ws;                        // 64 KB
    bf16_t* Abuf   = (bf16_t*)((char*)d_ws + 64 * 1024);  // 20 MB

    gate_kernel<<<T_TOK / 8, 256, 0, stream>>>(x, gate_w, gate_bias, mask_w);

    // routed experts: 32 M-tiles * 8 experts * 4 N-tiles
    moe_gateup_kernel<false><<<32 * 32, 256, 0, stream>>>(x, w_gate, w_up, mask_w, Abuf);
    // shared expert: 32 M-tiles * 8 N-tiles
    moe_gateup_kernel<true><<<32 * 8, 256, 0, stream>>>(x, sw_gate, sw_up, mask_w, Abuf);

    moe_down_kernel<<<32 * 8, 256, 0, stream>>>(Abuf, w_down, sw_down, (float*)d_out);
}
